// Decoder_61744449847461
// MI455X (gfx1250) — matrix operand order repaired and checked by reading, not compiled, not run on hardware
//
#include <hip/hip_runtime.h>
#include <hip/hip_bf16.h>
#include <math.h>

typedef __attribute__((ext_vector_type(16))) _Float16 v16h;
typedef __attribute__((ext_vector_type(8)))  float    v8f;
typedef __attribute__((ext_vector_type(4)))  float    v4f;

#define H_DIM 1024

// ---------------------------------------------------------------------------
// Async global->LDS staging (ASYNCcnt-tracked).  issue() may be called
// multiple times; finish() waits for all outstanding async copies and
// barriers the block.
// ---------------------------------------------------------------------------
__device__ __forceinline__ void stage_issue(float* dstLDS, const float* src,
                                            int nFloats) {
  const unsigned base = (unsigned)(uintptr_t)dstLDS;
  const unsigned long long gbase = (unsigned long long)(uintptr_t)src;
  for (int off = threadIdx.x * 16; off < nFloats * 4; off += blockDim.x * 16) {
    unsigned lds = base + (unsigned)off;
    unsigned long long g = gbase + (unsigned long long)off;
    asm volatile("global_load_async_to_lds_b128 %0, %1, off"
                 :: "v"(lds), "v"(g) : "memory");
  }
}

__device__ __forceinline__ void stage_finish() {
  asm volatile("s_wait_asynccnt 0" ::: "memory");
  __syncthreads();
}

// WGP-scope prefetch: fills all cache levels (near cache + GL2).  The builtin
// only reaches SE/SYS scopes, which skip the WGP cache, so use asm (default
// scope == 0 == WGP).
__device__ __forceinline__ void prefetch_wgp(const float* p) {
  asm volatile("global_prefetch_b8 %0, off" :: "v"((unsigned long long)(uintptr_t)p));
}

// ---------------------------------------------------------------------------
// f32 quads -> f16 x16 (A/B operand register image)
// ---------------------------------------------------------------------------
__device__ __forceinline__ v16h cvt16(v4f a0, v4f a1, v4f a2, v4f a3) {
  v16h r;
#pragma unroll
  for (int i = 0; i < 4; ++i) {
    r[i]      = (_Float16)a0[i];
    r[i + 4]  = (_Float16)a1[i];
    r[i + 8]  = (_Float16)a2[i];
    r[i + 12] = (_Float16)a3[i];
  }
  return r;
}

// A tile slice for this lane: cols (k..k+7) and (k+16..k+23) relative to the
// lane's group offset (already folded into the pointer).
__device__ __forceinline__ v16h loadA(const float* __restrict__ p) {
  return cvt16(*(const v4f*)(p), *(const v4f*)(p + 4),
               *(const v4f*)(p + 16), *(const v4f*)(p + 20));
}

// B slice: activations (k..k+7) and (k+16..k+23) (LDS-resident).
__device__ __forceinline__ v16h loadB(const float* p) {
  return cvt16(*(const v4f*)(p), *(const v4f*)(p + 4),
               *(const v4f*)(p + 16), *(const v4f*)(p + 20));
}

// ---------------------------------------------------------------------------
// One wave computes 16 consecutive rows of  out = W * vec  (row-major W, row
// stride ld) using V_WMMA_F32_16X16X32_F16, f32 accumulate.  W streams from
// global (L2-resident weights); vec comes from an LDS staging buffer.
//
// A layout (16-bit A 16x32, wave32):
//   lane L (0..15):  row L,     elems 0..7 = K k+0..7,   8..15 = K k+16..23
//   lane L (16..31): row L-16,  elems 0..7 = K k+8..15,  8..15 = K k+24..31
// B layout (16-bit B 32x16), column N=0 only:
//   lane 0 = K k+0..7,16..23 ; lane 16 = K k+8..15,24..31 ; all other lanes zero.
// Dual accumulators break the WMMA->WMMA D/C RAW chain (5-slot f16 hazard).
//
// D column 0: lane 0 VGPR r = row0+r (r=0..7); lane 16 VGPR r = row0+8+r.
//
// pfOK: enable the +2KB lookahead prefetch (disabled for the final row-block
// so the WGP-scope/non-speculative prefetch never walks past the matrix).
// ---------------------------------------------------------------------------
__device__ __forceinline__ v8f wmma_mv_acc(v8f acc,
                                           const float* __restrict__ W, int ld,
                                           int row0,
                                           const float* vec, int K,
                                           bool pfOK) {
  const int lane   = threadIdx.x & 31;
  const int laneLo = lane & 15;
  const int grp    = lane >> 4;
  const float* wrow = W + (size_t)(row0 + laneLo) * (size_t)ld + grp * 8;
  const float* vrow = vec + grp * 8;
  const float* pfrow = pfOK ? (wrow + 512) : wrow;   // hoisted select
  const bool isCol0 = (laneLo == 0);
  const v16h Bz = {};
  v8f acc1 = {};

#pragma unroll 2
  for (int k = 0; k < K; k += 64) {
    // Weights live in L2 after the first step; pull ~2KB ahead into the
    // WGP near cache.
    prefetch_wgp(pfrow + k);

    // ---- chunk 0: K = k .. k+31 ----
    v16h A0 = loadA(wrow + k);
    v16h B0 = isCol0 ? loadB(vrow + k) : Bz;
    acc  = __builtin_amdgcn_wmma_f32_16x16x32_f16(false, A0, false, B0,
                                                  (short)0, acc,  false, false);
    // ---- chunk 1: K = k+32 .. k+63 ----
    v16h A1 = loadA(wrow + k + 32);
    v16h B1 = isCol0 ? loadB(vrow + k + 32) : Bz;
    acc1 = __builtin_amdgcn_wmma_f32_16x16x32_f16(false, A1, false, B1,
                                                  (short)0, acc1, false, false);
  }
#pragma unroll
  for (int i = 0; i < 8; ++i) acc[i] += acc1[i];
  return acc;
}

__device__ __forceinline__ void store_rows16(v8f acc, float* __restrict__ out,
                                             int row0) {
  const int lane = threadIdx.x & 31;
  if (lane == 0) {
#pragma unroll
    for (int i = 0; i < 8; ++i) out[row0 + i] = acc[i];
  } else if (lane == 16) {
#pragma unroll
    for (int i = 0; i < 8; ++i) out[row0 + 8 + i] = acc[i];
  }
}

__device__ __forceinline__ float sigmoidf_(float x) {
  return 1.0f / (1.0f + __expf(-x));
}

// ---------------------------------------------------------------------------
// LSTM layer: gates = Wih*x + Whh*h  (4096 rows, K=1024 each), then the last
// arriving block applies biases + pointwise LSTM update -> hstate, cstate.
// grid = 64 blocks * 128 threads (4 waves) -> 256 waves, 16 rows each.
// ---------------------------------------------------------------------------
__global__ void __launch_bounds__(128)
lstm_layer_kernel(const float* __restrict__ Wih, const float* __restrict__ Whh,
                  const float* __restrict__ xvec,
                  const float* __restrict__ bih, const float* __restrict__ bhh,
                  float* __restrict__ hstate, float* __restrict__ cstate,
                  float* __restrict__ gates, unsigned* __restrict__ cnt) {
  __shared__ float lvec[2 * H_DIM];   // [x | h]
  stage_issue(lvec, xvec, H_DIM);
  stage_issue(lvec + H_DIM, hstate, H_DIM);
  stage_finish();

  const int wave = threadIdx.x >> 5;
  const int row0 = (blockIdx.x * 4 + wave) * 16;
  const bool pfOK = row0 + 16 < 4096;

  v8f acc = {};
  acc = wmma_mv_acc(acc, Wih, H_DIM, row0, lvec, H_DIM, true);  // Whh follows
  acc = wmma_mv_acc(acc, Whh, H_DIM, row0, lvec + H_DIM, H_DIM, pfOK);
  store_rows16(acc, gates, row0);

  __shared__ int isLast;
  __threadfence();
  __syncthreads();
  if (threadIdx.x == 0) {
    unsigned p = atomicAdd(cnt, 1u);
    isLast = (p == gridDim.x - 1) ? 1 : 0;
  }
  __syncthreads();
  if (isLast) {
    for (int j = threadIdx.x; j < H_DIM; j += 128) {
      float gi = gates[j]            + bih[j]            + bhh[j];
      float gf = gates[H_DIM + j]    + bih[H_DIM + j]    + bhh[H_DIM + j];
      float gg = gates[2*H_DIM + j]  + bih[2*H_DIM + j]  + bhh[2*H_DIM + j];
      float go = gates[3*H_DIM + j]  + bih[3*H_DIM + j]  + bhh[3*H_DIM + j];
      float c_ = sigmoidf_(gf) * cstate[j] + sigmoidf_(gi) * tanhf(gg);
      cstate[j] = c_;
      hstate[j] = sigmoidf_(go) * tanhf(c_);
    }
    __threadfence();
    if (threadIdx.x == 0) *cnt = 0u;
  }
}

// ---------------------------------------------------------------------------
// Attention scores: scores[s] = enc[s,:] . h1  (1024 rows, K=1024), then last
// block does a numerically-stable softmax over the 1024 scores -> wght.
// grid = 16 blocks * 128 threads -> 64 waves.
// ---------------------------------------------------------------------------
__global__ void __launch_bounds__(128)
attn_scores_kernel(const float* __restrict__ enc, const float* __restrict__ h1,
                   float* __restrict__ scores, float* __restrict__ wght,
                   unsigned* __restrict__ cnt) {
  __shared__ float lvec[H_DIM];
  stage_issue(lvec, h1, H_DIM);
  stage_finish();

  const int wave = threadIdx.x >> 5;
  const int row0 = (blockIdx.x * 4 + wave) * 16;
  const bool pfOK = row0 + 16 < 1024;

  v8f acc = {};
  acc = wmma_mv_acc(acc, enc, H_DIM, row0, lvec, H_DIM, pfOK);
  store_rows16(acc, scores, row0);

  __shared__ int isLast;
  __shared__ float red[128];
  __threadfence();
  __syncthreads();
  if (threadIdx.x == 0) {
    unsigned p = atomicAdd(cnt, 1u);
    isLast = (p == gridDim.x - 1) ? 1 : 0;
  }
  __syncthreads();
  if (isLast) {
    float m = -3.4e38f;
    for (int j = threadIdx.x; j < H_DIM; j += 128) m = fmaxf(m, scores[j]);
    red[threadIdx.x] = m;
    __syncthreads();
    for (int s = 64; s > 0; s >>= 1) {
      if (threadIdx.x < s)
        red[threadIdx.x] = fmaxf(red[threadIdx.x], red[threadIdx.x + s]);
      __syncthreads();
    }
    float smax = red[0];
    __syncthreads();
    float sum = 0.f;
    for (int j = threadIdx.x; j < H_DIM; j += 128) {
      float e = __expf(scores[j] - smax);
      wght[j] = e;
      sum += e;
    }
    red[threadIdx.x] = sum;
    __syncthreads();
    for (int s = 64; s > 0; s >>= 1) {
      if (threadIdx.x < s) red[threadIdx.x] += red[threadIdx.x + s];
      __syncthreads();
    }
    float inv = 1.0f / red[0];
    for (int j = threadIdx.x; j < H_DIM; j += 128) wght[j] *= inv;
    __threadfence();
    if (threadIdx.x == 0) *cnt = 0u;
  }
}

// ---------------------------------------------------------------------------
// Context: ctx[h] = sum_s enc[s,h] * wght[s] — column access.  4 columns per
// thread (b128 loads), weights async-staged to LDS.  grid = 2 * 128.
// ---------------------------------------------------------------------------
__global__ void __launch_bounds__(128)
attn_context_kernel(const float* __restrict__ enc,
                    const float* __restrict__ wght, float* __restrict__ ctx) {
  __shared__ float lw[H_DIM];
  stage_issue(lw, wght, H_DIM);
  stage_finish();

  const int col4 = (blockIdx.x * 128 + threadIdx.x) * 4;   // 0..1020
  v4f acc = {};
#pragma unroll 4
  for (int s = 0; s < H_DIM; ++s) {
    v4f e = *(const v4f*)(enc + (size_t)s * H_DIM + col4);
    float w = lw[s];
#pragma unroll
    for (int i = 0; i < 4; ++i) acc[i] = fmaf(e[i], w, acc[i]);
  }
  *(v4f*)(ctx + col4) = acc;
}

// ---------------------------------------------------------------------------
// attn_h = tanh(Wc * cat + bc), cat = [context ; h1] (K=2048, 1024 rows).
// Writes the timestep output slice and the next-step x.  16 blocks * 128.
// ---------------------------------------------------------------------------
__global__ void __launch_bounds__(128)
concat_out_kernel(const float* __restrict__ Wc, const float* __restrict__ cat,
                  const float* __restrict__ bc, float* __restrict__ outT,
                  float* __restrict__ xbuf) {
  __shared__ float lvec[2 * H_DIM];
  stage_issue(lvec, cat, 2 * H_DIM);
  stage_finish();

  const int wave = threadIdx.x >> 5;
  const int row0 = (blockIdx.x * 4 + wave) * 16;
  const bool pfOK = row0 + 16 < 1024;

  v8f acc = {};
  acc = wmma_mv_acc(acc, Wc, 2 * H_DIM, row0, lvec, 2 * H_DIM, pfOK);

  const int lane = threadIdx.x & 31;
  if (lane == 0) {
#pragma unroll
    for (int i = 0; i < 8; ++i) {
      int r = row0 + i;
      float v = tanhf(acc[i] + bc[r]);
      outT[r] = v;
      xbuf[r] = v;
    }
  } else if (lane == 16) {
#pragma unroll
    for (int i = 0; i < 8; ++i) {
      int r = row0 + 8 + i;
      float v = tanhf(acc[i] + bc[r]);
      outT[r] = v;
      xbuf[r] = v;
    }
  }
}

// ---------------------------------------------------------------------------
// Per-call state init (deterministic, graph-capture safe).
// ---------------------------------------------------------------------------
__global__ void __launch_bounds__(128)
init_kernel(const float* __restrict__ hidden, const float* __restrict__ cell,
            float* __restrict__ xbuf, float* __restrict__ h0,
            float* __restrict__ cat, float* __restrict__ c0,
            float* __restrict__ c1, unsigned* __restrict__ cnt) {
  const int j = blockIdx.x * 128 + threadIdx.x;
  if (j < H_DIM) {
    h0[j] = hidden[j];               // layer 0 h
    float h1v = hidden[H_DIM + j];   // layer 1 h (== hidden_state[-1])
    xbuf[j] = h1v;                   // initial x = hidden_state[-1]
    cat[H_DIM + j] = h1v;            // h1 lives in cat[1024..2047]
    c0[j] = cell[j];
    c1[j] = cell[H_DIM + j];
  }
  if (j == 0) *cnt = 0u;
}

extern "C" void kernel_launch(void* const* d_in, const int* in_sizes, int n_in,
                              void* d_out, int out_size, void* d_ws,
                              size_t ws_size, hipStream_t stream) {
  const float* enc      = (const float*)d_in[1];  // [1024, 1024]
  const float* hidden   = (const float*)d_in[2];  // [2, 1, 1024]
  const float* cell     = (const float*)d_in[3];  // [2, 1, 1024]
  const float* w_ih     = (const float*)d_in[4];  // [2, 4096, 1024]
  const float* w_hh     = (const float*)d_in[5];  // [2, 4096, 1024]
  const float* b_ih     = (const float*)d_in[6];  // [2, 4096]
  const float* b_hh     = (const float*)d_in[7];  // [2, 4096]
  const float* w_concat = (const float*)d_in[8];  // [1024, 2048]
  const float* b_concat = (const float*)d_in[9];  // [1024]
  float* out = (float*)d_out;                     // [T, 1, 1024]

  const int T = in_sizes[0] / H_DIM;              // 256

  // workspace carve (all chunks 4KB-multiple offsets -> aligned)
  float* xbuf   = (float*)d_ws;        // 1024
  float* h0     = xbuf + 1024;         // 1024
  float* cat    = h0 + 1024;           // 2048  = [ctx | h1]
  float* c0     = cat + 2048;          // 1024
  float* c1     = c0 + 1024;           // 1024
  float* gates  = c1 + 1024;           // 4096
  float* scores = gates + 4096;        // 1024
  float* wght   = scores + 1024;       // 1024
  unsigned* cnt = (unsigned*)(wght + 1024);

  const size_t wslab = (size_t)4096 * 1024;  // per-layer weight block

  init_kernel<<<8, 128, 0, stream>>>(hidden, cell, xbuf, h0, cat, c0, c1, cnt);

  for (int t = 0; t < T; ++t) {
    // layer 0: x = xbuf, h = h0
    lstm_layer_kernel<<<64, 128, 0, stream>>>(w_ih, w_hh, xbuf,
                                              b_ih, b_hh, h0, c0, gates, cnt);
    // layer 1: x = h0, h = h1 (stored at cat+1024)
    lstm_layer_kernel<<<64, 128, 0, stream>>>(w_ih + wslab, w_hh + wslab, h0,
                                              b_ih + 4096, b_hh + 4096,
                                              cat + H_DIM, c1, gates, cnt);
    // attention
    attn_scores_kernel<<<16, 128, 0, stream>>>(enc, cat + H_DIM, scores, wght,
                                               cnt);
    attn_context_kernel<<<2, 128, 0, stream>>>(enc, wght, cat);
    // output projection (+ feeds next step's x)
    concat_out_kernel<<<16, 128, 0, stream>>>(w_concat, cat, b_concat,
                                              out + (size_t)t * H_DIM, xbuf);
  }
}
